// Rnn_12816182411522
// MI455X (gfx1250) — compile-verified, matrix operand order checked by reading, not run on hardware
//
#include <hip/hip_runtime.h>

typedef _Float16 v16h  __attribute__((ext_vector_type(16)));
typedef float    v8f   __attribute__((ext_vector_type(8)));
typedef float    f32x4 __attribute__((ext_vector_type(4)));
typedef _Float16 h2    __attribute__((ext_vector_type(2)));

#define T_LEN 512
#define B_SZ  1024
#define I_SZ  64
#define H_SZ  32
#define PF_AHEAD 12

__device__ __forceinline__ float fast_tanh(float xx) {
#if __has_builtin(__builtin_amdgcn_tanhf)
  return __builtin_amdgcn_tanhf(xx);
#else
  // tanh(x) = 1 - 2/(exp(2x)+1); v_exp_f32 is exp2
  float e = __builtin_amdgcn_exp2f(xx * 2.8853900817779268f); // 2*log2(e)
  return 1.0f - 2.0f * __builtin_amdgcn_rcpf(e + 1.0f);
#endif
}

// v_cvt_pk_rtz_f16_f32 wrapper: returns a _Float16x2 via bit_cast (builtin's
// native return type is __fp16x2, same layout).
__device__ __forceinline__ h2 pk_f16(float a, float b) {
  auto p = __builtin_amdgcn_cvt_pkrtz(a, b);
  return __builtin_bit_cast(h2, p);
}

// Pack 8 f32 (D-fragment regs) into 8 f16 halves of a B-fragment, lane-local.
__device__ __forceinline__ void pack8(v16h& frag, int base, const v8f& v) {
#pragma unroll
  for (int j = 0; j < 4; ++j) {
    h2 p = pk_f16(v[2 * j], v[2 * j + 1]);
    frag[base + 2 * j]     = p.x;
    frag[base + 2 * j + 1] = p.y;
  }
}

// One wave per 16-row batch tile. D = A(W) x B(h/x): M=n(hidden), N=b(batch).
// A-frag (16-bit 16x32): lane m=lane&15; halves j<8 -> K=kA+j, j>=8 -> K=16+kA+(j-8), kA=8*(lane>>4).
// B-frag mirrors with N=lane&15. D-frag: lane=col b; VGPR r -> n = r + 8*(lane>>4).
// D(t) -> B(t+1) is therefore lane-local: B[0..7]=cvt(D0[0..7]), B[8..15]=cvt(D1[0..7]).
__global__ __launch_bounds__(64) void rnn_fused_kernel(
    const float* __restrict__ x,     const float* __restrict__ h0,
    const float* __restrict__ W_ih,  const float* __restrict__ b_ih,
    const float* __restrict__ W_hh,  const float* __restrict__ b_hh,
    const float* __restrict__ W_out, const float* __restrict__ b_out,
    float* __restrict__ out) {
  const int lane  = threadIdx.x & 31;
  const int tile  = blockIdx.x * (blockDim.x >> 5) + (threadIdx.x >> 5); // 0..63
  const int bBase = tile * 16;
  const int col   = lane & 15;
  const int hi    = lane >> 4;
  const int kA    = hi * 8;

  // ---- constant A fragments: W_ih (2 n-tiles x 2 K-blocks), W_hh (2 n-tiles) ----
  v16h Aih[2][2];
#pragma unroll
  for (int nt = 0; nt < 2; ++nt) {
#pragma unroll
    for (int kb = 0; kb < 2; ++kb) {
      const int row = nt * 16 + col;
#pragma unroll
      for (int j = 0; j < 16; ++j) {
        const int K = (j < 8) ? (kA + j) : (16 + kA + (j - 8));
        Aih[nt][kb][j] = (_Float16)W_ih[row * I_SZ + kb * 32 + K];
      }
    }
  }
  v16h Ahh[2];
#pragma unroll
  for (int nt = 0; nt < 2; ++nt) {
    const int row = nt * 16 + col;
#pragma unroll
    for (int j = 0; j < 16; ++j) {
      const int K = (j < 8) ? (kA + j) : (16 + kA + (j - 8));
      Ahh[nt][j] = (_Float16)W_hh[row * H_SZ + K];
    }
  }
  // ---- bias as C fragments; W_out in D layout for the output dot ----
  v8f   cbias[2];
  float wout[2][8];
#pragma unroll
  for (int nt = 0; nt < 2; ++nt) {
#pragma unroll
    for (int r = 0; r < 8; ++r) {
      const int n = nt * 16 + 8 * hi + r;
      cbias[nt][r] = b_ih[n] + b_hh[n];
      wout[nt][r]  = W_out[n];
    }
  }
  const float bout = b_out[0];

  // ---- initial hidden state as B fragment ----
  v16h hfrag;
  {
    const float* hp = h0 + (size_t)(bBase + col) * H_SZ;
#pragma unroll
    for (int j = 0; j < 16; ++j) {
      const int K = (j < 8) ? (kA + j) : (16 + kA + (j - 8));
      hfrag[j] = (_Float16)hp[K];
    }
  }

  const float* rowPtr = x + (size_t)(bBase + col) * T_LEN * I_SZ;

  auto loadRaw = [&](int t, f32x4* L) {
    const float* p = rowPtr + (size_t)t * I_SZ;
#pragma unroll
    for (int f = 0; f < 2; ++f) {
      const float* q = p + f * 32 + kA;
      L[f * 4 + 0] = *(const f32x4*)(q);
      L[f * 4 + 1] = *(const f32x4*)(q + 4);
      L[f * 4 + 2] = *(const f32x4*)(q + 16);
      L[f * 4 + 3] = *(const f32x4*)(q + 20);
    }
  };
  auto cvtRaw = [&](const f32x4* L, v16h& f0, v16h& f1) {
#pragma unroll
    for (int f = 0; f < 2; ++f) {
      v16h& dst = f ? f1 : f0;
#pragma unroll
      for (int q = 0; q < 4; ++q) {
#pragma unroll
        for (int e = 0; e < 4; e += 2) {
          h2 p = pk_f16(L[f * 4 + q][e], L[f * 4 + q][e + 1]);
          const int j = q * 4 + e;
          dst[j] = p.x;
          dst[j + 1] = p.y;
        }
      }
    }
  };

  f32x4 rawL[8];
  v16h  bx0, bx1;
  loadRaw(0, rawL);
  cvtRaw(rawL, bx0, bx1);

  v8f h0f = {}, h1f = {};
  for (int t = 0; t < T_LEN; ++t) {
    // L2 warm-up for t+PF_AHEAD: 32 lanes cover the 16 rows x 256B tile.
    {
      int tp = t + PF_AHEAD;
      if (tp > T_LEN - 1) tp = T_LEN - 1;
      const float* pp = x + ((size_t)(bBase + (lane >> 1)) * T_LEN + tp) * I_SZ + (lane & 1) * 32;
      __builtin_prefetch(pp, 0, 0);
    }
    // Issue next step's loads now; convert at loop end (off critical path).
    int tn = t + 1;
    if (tn > T_LEN - 1) tn = T_LEN - 1;
    f32x4 nxt[8];
    loadRaw(tn, nxt);

    // x-projection (independent of h) then recurrence on the serial chain.
    v8f d0 = cbias[0], d1 = cbias[1];
    d0 = __builtin_amdgcn_wmma_f32_16x16x32_f16(false, Aih[0][0], false, bx0, (short)0, d0, false, false);
    d0 = __builtin_amdgcn_wmma_f32_16x16x32_f16(false, Aih[0][1], false, bx1, (short)0, d0, false, false);
    d1 = __builtin_amdgcn_wmma_f32_16x16x32_f16(false, Aih[1][0], false, bx0, (short)0, d1, false, false);
    d1 = __builtin_amdgcn_wmma_f32_16x16x32_f16(false, Aih[1][1], false, bx1, (short)0, d1, false, false);
    d0 = __builtin_amdgcn_wmma_f32_16x16x32_f16(false, Ahh[0], false, hfrag, (short)0, d0, false, false);
    d1 = __builtin_amdgcn_wmma_f32_16x16x32_f16(false, Ahh[1], false, hfrag, (short)0, d1, false, false);

#pragma unroll
    for (int r = 0; r < 8; ++r) {
      h0f[r] = fast_tanh(d0[r]);
      h1f[r] = fast_tanh(d1[r]);
    }

    // D(t) -> B(t+1), lane-local.
    pack8(hfrag, 0, h0f);
    pack8(hfrag, 8, h1f);

    // outs[b,t] = h . W_out + b_out ; lane and lane^16 each hold 16 of 32 n-values.
    float partial = 0.f;
#pragma unroll
    for (int r = 0; r < 8; ++r)
      partial += h0f[r] * wout[0][r] + h1f[r] * wout[1][r];
    partial += __shfl_xor(partial, 16, 32);
    if (lane < 16) out[(size_t)(bBase + col) * T_LEN + t] = partial + bout;

    cvtRaw(nxt, bx0, bx1);
  }

  // h_last: [1, B, H] appended after outs [B, T].
  float* hl = out + (size_t)B_SZ * T_LEN;
#pragma unroll
  for (int r = 0; r < 8; ++r) {
    const int n = 8 * hi + r;
    hl[(size_t)(bBase + col) * H_SZ + n]      = h0f[r];
    hl[(size_t)(bBase + col) * H_SZ + n + 16] = h1f[r];
  }
}

extern "C" void kernel_launch(void* const* d_in, const int* in_sizes, int n_in,
                              void* d_out, int out_size, void* d_ws, size_t ws_size,
                              hipStream_t stream) {
  const float* x     = (const float*)d_in[0];
  const float* h0    = (const float*)d_in[1];
  const float* W_ih  = (const float*)d_in[2];
  const float* b_ih  = (const float*)d_in[3];
  const float* W_hh  = (const float*)d_in[4];
  const float* b_hh  = (const float*)d_in[5];
  const float* W_out = (const float*)d_in[6];
  const float* b_out = (const float*)d_in[7];
  float* out = (float*)d_out;

  // 64 batch tiles, 1 wave each; 2 waves per block -> 32 blocks spread across WGPs.
  dim3 grid(32), block(64);
  hipLaunchKernelGGL(rnn_fused_kernel, grid, block, 0, stream,
                     x, h0, W_ih, b_ih, W_hh, b_hh, W_out, b_out, out);
}
